// upblock_36524401885205
// MI455X (gfx1250) — compile-verified, matrix operand order checked by reading, not run on hardware
//
#include <hip/hip_runtime.h>
#include <hip/hip_bf16.h>

// ---------------------------------------------------------------------------
// Sparse up-block (deconv 128->128, concat 64, conv 192->96) via bf16 WMMA.
// One wave32 handles one 16-row edge tile; accumulators are f32 16x16 tiles.
// ---------------------------------------------------------------------------

typedef __attribute__((ext_vector_type(16))) __bf16        v16bf;
typedef __attribute__((ext_vector_type(8)))  float         v8f;
typedef __attribute__((ext_vector_type(4)))  unsigned int  u32x4;

#define N_DOWN 40000
#define N_UP   160000
#define C_DOWN 128
#define C_SKIP 64
#define C_CAT  192
#define C_OUT  96
#define KOFF   27

union Frag { v16bf v; u32x4 q[2]; };

__device__ __forceinline__ unsigned short f2bf(float f) {
  unsigned int x = __float_as_uint(f);
  x += 0x7fffu + ((x >> 16) & 1u);          // round-to-nearest-even
  return (unsigned short)(x >> 16);
}

__device__ __forceinline__ void atomAddF32(float* p, float v) {
  (void)__hip_atomic_fetch_add(p, v, __ATOMIC_RELAXED, __HIP_MEMORY_SCOPE_AGENT);
}

// ---------------------------- elementwise kernels ---------------------------

__global__ void zero_f32(float* __restrict__ p, int n) {
  int i = blockIdx.x * blockDim.x + threadIdx.x;
  if (i < n) p[i] = 0.0f;
}

__global__ void relu_inplace(float* __restrict__ p, int n) {
  int i = blockIdx.x * blockDim.x + threadIdx.x;
  if (i < n) { float v = p[i]; p[i] = v > 0.0f ? v : 0.0f; }
}

__global__ void prep_down(const float* __restrict__ src,
                          unsigned short* __restrict__ dst, int n) {
  int i = blockIdx.x * blockDim.x + threadIdx.x;
  if (i < n) dst[i] = f2bf(src[i]);
}

// skip fp32 -> bf16 into cat columns [128..192)
__global__ void prep_skip(const float* __restrict__ skip,
                          unsigned short* __restrict__ cat) {
  int i = blockIdx.x * blockDim.x + threadIdx.x;
  if (i < N_UP * C_SKIP) {
    int row = i / C_SKIP, c = i - row * C_SKIP;
    cat[(size_t)row * C_CAT + C_DOWN + c] = f2bf(skip[i]);
  }
}

// ReLU(up_acc) fp32 -> bf16 into cat columns [0..128)
__global__ void relu_cat(const float* __restrict__ up,
                         unsigned short* __restrict__ cat) {
  int i = blockIdx.x * blockDim.x + threadIdx.x;
  if (i < N_UP * C_DOWN) {
    int row = i >> 7, c = i & 127;
    float v = up[i];
    v = v > 0.0f ? v : 0.0f;
    cat[(size_t)row * C_CAT + c] = f2bf(v);
  }
}

// Swizzle W[k][kk][n] (fp32) into per-lane WMMA B fragments (bf16):
// dst index = (((k*KT + kt)*NT + nt)*32 + lane)*16 + e
// lane holds column n = nt*16 + (lane&15); K layout matches the 16-bit WMMA
// operand layout: lane<16 -> K {0..7,16..23}, lane>=16 -> K {8..15,24..31}.
__global__ void prep_w(const float* __restrict__ w,
                       unsigned short* __restrict__ frag,
                       int cin, int cout, int KT, int NT, int total) {
  int i = blockIdx.x * blockDim.x + threadIdx.x;
  if (i >= total) return;
  int e = i & 15;
  int t = i >> 4;
  int lane = t & 31; t >>= 5;
  int nt = t % NT; t /= NT;
  int kt = t % KT;
  int k  = t / KT;
  int n  = nt * 16 + (lane & 15);
  int kb = (lane >> 4) << 3;
  int koff = (e < 8) ? (kb + e) : (kb + 16 + (e - 8));
  int kk = kt * 32 + koff;
  frag[i] = f2bf(w[((size_t)k * cin + kk) * cout + n]);
}

// ------------------------------ WMMA sparse conv ----------------------------
// featb : bf16 feature rows [*, CIN]
// wfrag : pre-swizzled bf16 weight fragments
// One wave = 16 edges of one kernel offset k; KT = CIN/32 K-steps, NT = COUT/16
// output tiles; scatter-add f32 into out_acc.
template <int KT, int NT, int CIN, int COUT, int NE>
__global__ void __launch_bounds__(256) spconv_wmma(
    const unsigned short* __restrict__ featb,
    const unsigned short* __restrict__ wfrag,
    const int* __restrict__ in_idx,
    const int* __restrict__ out_idx,
    float* __restrict__ out_acc) {
  const int ET = NE / 16;                         // edge tiles per offset
  int wave = blockIdx.x * (blockDim.x >> 5) + (threadIdx.x >> 5);
  if (wave >= KOFF * ET) return;                  // wave-uniform: EXEC stays all-1
  int lane = threadIdx.x & 31;
  int k  = wave / ET;
  int e0 = (wave - k * ET) * 16;

  const int* ii = in_idx  + (size_t)k * NE + e0;
  const int* oi = out_idx + (size_t)k * NE + e0;

  // A operand: row m = lane&15 of the gathered tile, K chunk split by lane half
  int arow = ii[lane & 15];
  const u32x4* abase =
      (const u32x4*)(featb + (size_t)arow * CIN + ((lane >> 4) << 3));
  // B operand: fragments laid out contiguously per (kt,nt), 16 bf16 per lane
  const u32x4* wbase =
      (const u32x4*)(wfrag + (size_t)k * KT * NT * 512) + lane * 2;

  v8f acc[NT];
  const v8f vzero = {0, 0, 0, 0, 0, 0, 0, 0};
#pragma unroll
  for (int nt = 0; nt < NT; ++nt) acc[nt] = vzero;

#pragma unroll
  for (int kt = 0; kt < KT; ++kt) {
    Frag a;
    a.q[0] = abase[kt * 4];          // K offsets {0..7} (+8 for upper lanes)
    a.q[1] = abase[kt * 4 + 2];      // K offsets {16..23} (+8 for upper lanes)
#pragma unroll
    for (int nt = 0; nt < NT; ++nt) {
      Frag b;
      const u32x4* bp = wbase + (kt * NT + nt) * 64;
      b.q[0] = bp[0];
      b.q[1] = bp[1];
      acc[nt] = __builtin_amdgcn_wmma_f32_16x16x32_bf16(
          false, a.v, false, b.v, (short)0, acc[nt], false, false);
    }
  }

  // C/D layout: VGPR e -> row (e + 8*(lane>=16)), column lane&15 (+16*nt)
  int col = lane & 15;
  int mb  = (lane >> 4) << 3;
#pragma unroll
  for (int e = 0; e < 8; ++e) {
    int orow = oi[mb + e];
    float* base = out_acc + (size_t)orow * COUT + col;
#pragma unroll
    for (int nt = 0; nt < NT; ++nt) atomAddF32(base + nt * 16, acc[nt][e]);
  }
}

// --------------------------------- launcher ---------------------------------

extern "C" void kernel_launch(void* const* d_in, const int* in_sizes, int n_in,
                              void* d_out, int out_size, void* d_ws, size_t ws_size,
                              hipStream_t stream) {
  const float* skip = (const float*)d_in[0];
  const float* down = (const float*)d_in[1];
  const float* Wd   = (const float*)d_in[2];
  const float* Wc   = (const float*)d_in[3];
  const int* dii    = (const int*)d_in[4];
  const int* doi    = (const int*)d_in[5];
  const int* cii    = (const int*)d_in[6];
  const int* coi    = (const int*)d_in[7];
  float* out        = (float*)d_out;

  char* ws = (char*)d_ws;
  size_t off = 0;
  auto carve = [&](size_t bytes) -> char* {
    char* p = ws + off;
    off += (bytes + 255) & ~(size_t)255;
    return p;
  };
  float*          up_acc = (float*)carve((size_t)N_UP * C_DOWN * 4);   // 81.9 MB
  unsigned short* catb   = (unsigned short*)carve((size_t)N_UP * C_CAT * 2); // 61.4 MB
  unsigned short* downb  = (unsigned short*)carve((size_t)N_DOWN * C_DOWN * 2);
  unsigned short* wdf    = (unsigned short*)carve((size_t)KOFF * 4 * 8 * 512 * 2);
  unsigned short* wcf    = (unsigned short*)carve((size_t)KOFF * 6 * 6 * 512 * 2);

  const dim3 B(256);
  auto g = [](int n) { return dim3((unsigned)((n + 255) / 256)); };

  // 1) zero accumulators (out is also the conv accumulator)
  zero_f32<<<g(N_UP * C_DOWN), B, 0, stream>>>(up_acc, N_UP * C_DOWN);
  zero_f32<<<g(N_UP * C_OUT),  B, 0, stream>>>(out,    N_UP * C_OUT);

  // 2) bf16 conversions + weight fragment swizzles
  prep_down<<<g(N_DOWN * C_DOWN), B, 0, stream>>>(down, downb, N_DOWN * C_DOWN);
  prep_skip<<<g(N_UP * C_SKIP),   B, 0, stream>>>(skip, catb);
  prep_w<<<g(KOFF * 4 * 8 * 512), B, 0, stream>>>(Wd, wdf, C_DOWN, C_DOWN, 4, 8,
                                                  KOFF * 4 * 8 * 512);
  prep_w<<<g(KOFF * 6 * 6 * 512), B, 0, stream>>>(Wc, wcf, C_CAT, C_OUT, 6, 6,
                                                  KOFF * 6 * 6 * 512);

  // 3) deconv: 27 * 2500 tiles, 8 waves/block
  {
    int waves = KOFF * (N_DOWN / 16);
    spconv_wmma<4, 8, C_DOWN, C_DOWN, N_DOWN>
        <<<dim3((unsigned)((waves + 7) / 8)), B, 0, stream>>>(downb, wdf, dii, doi, up_acc);
  }

  // 4) ReLU + concat into bf16 cat buffer
  relu_cat<<<g(N_UP * C_DOWN), B, 0, stream>>>(up_acc, catb);

  // 5) conv: 27 * 10000 tiles, accumulate into d_out
  {
    int waves = KOFF * (N_UP / 16);
    spconv_wmma<6, 6, C_CAT, C_OUT, N_UP>
        <<<dim3((unsigned)((waves + 7) / 8)), B, 0, stream>>>(catb, wcf, cii, coi, out);
  }

  // 6) final ReLU in place
  relu_inplace<<<g(N_UP * C_OUT), B, 0, stream>>>(out, N_UP * C_OUT);
}
